// MambaBlock_82420422410414
// MI455X (gfx1250) — matrix operand order repaired and checked by reading, not compiled, not run on hardware
//
#include <hip/hip_runtime.h>
#include <hip/hip_bf16.h>
#include <math.h>

#define D_MODEL 1024
#define D_STATE 16
#define D_INNER 2048
#define BATCH   4
#define SEQ     2048
#define LN_EPS  1e-5f
#define MROWS   (BATCH * SEQ)   /* 8192 rows for all GEMMs */
#define NCH     16              /* scan time-chunks */
#define CHUNK   (SEQ / NCH)     /* 128 steps per chunk */

typedef __attribute__((ext_vector_type(16))) __bf16 v16bf;
typedef __attribute__((ext_vector_type(8)))  float  v8f;

__device__ __forceinline__ unsigned short f2bf(float f) {
    unsigned int u = __float_as_uint(f);
    u += 0x7fffu + ((u >> 16) & 1u);   // round-to-nearest-even
    return (unsigned short)(u >> 16);
}
__device__ __forceinline__ float sigmf(float x) { return 1.0f / (1.0f + __expf(-x)); }

// ---------------------------------------------------------------------------
// Repack weight W[K,N] (fp32, row-major) into bf16 B-fragment tiles.
// Tile (kt,nt): 512 elems, lane-major; lane L -> N=nt*16+(L&15), h=L>>4, slot s ->
// K=kt*32+8h+s (s<8), K=kt*32+16+8h+(s-8) (s>=8). Lane fragment = 32B contig.
// ---------------------------------------------------------------------------
__global__ __launch_bounds__(256) void repack_w(const float* __restrict__ W,
                                                unsigned short* __restrict__ Wp,
                                                int K, int N) {
    int idx = blockIdx.x * 256 + threadIdx.x;
    if (idx >= K * N) return;
    int e    = idx & 511;
    int lane = e >> 4, s = e & 15;
    int tile = idx >> 9;
    int Nt   = N >> 4;
    int nt   = tile % Nt, kt = tile / Nt;
    int k = (kt << 5) + ((s >> 3) << 4) + ((lane >> 4) << 3) + (s & 7);
    int n = (nt << 4) + (lane & 15);
    Wp[idx] = f2bf(W[(size_t)k * N + n]);
}

// ---------------------------------------------------------------------------
// LayerNorm over d_model, writing xn directly in bf16 A-fragment layout.
// A tile (mt,kt): half h (0: K in {0..7,16..23}; 1: K in {8..15,24..31}),
// row r = m&15, slot s; offset = (mt*Ktiles+kt)*512 + h*256 + r*16 + s.
// ---------------------------------------------------------------------------
__global__ __launch_bounds__(256) void ln_pack(const float* __restrict__ x,
                                               const float* __restrict__ gamma,
                                               const float* __restrict__ beta,
                                               unsigned short* __restrict__ xp) {
    int m   = blockIdx.x;
    int tid = threadIdx.x;
    const float4 v = reinterpret_cast<const float4*>(x + (size_t)m * D_MODEL)[tid];
    float s  = v.x + v.y + v.z + v.w;
    float sq = v.x * v.x + v.y * v.y + v.z * v.z + v.w * v.w;
    #pragma unroll
    for (int off = 16; off > 0; off >>= 1) {
        s  += __shfl_down(s,  off, 32);
        sq += __shfl_down(sq, off, 32);
    }
    __shared__ float ws1[8], ws2[8];
    int wv = tid >> 5, ln = tid & 31;
    if (ln == 0) { ws1[wv] = s; ws2[wv] = sq; }
    __syncthreads();
    if (tid == 0) {
        float a = 0.f, b = 0.f;
        for (int i = 0; i < 8; ++i) { a += ws1[i]; b += ws2[i]; }
        ws1[0] = a; ws2[0] = b;
    }
    __syncthreads();
    float mu   = ws1[0] * (1.0f / D_MODEL);
    float var  = ws2[0] * (1.0f / D_MODEL) - mu * mu;
    float rstd = rsqrtf(var + LN_EPS);

    int r = m & 15;
    size_t base = ((size_t)(m >> 4) * (D_MODEL / 32)) << 9;
    #pragma unroll
    for (int j = 0; j < 4; ++j) {
        int   k   = tid * 4 + j;
        float val = (j == 0) ? v.x : (j == 1) ? v.y : (j == 2) ? v.z : v.w;
        float xn  = (val - mu) * rstd * gamma[k] + beta[k];
        int kt = k >> 5, kk = k & 31;
        int half = (kk >> 3) & 1;
        int slot = (kk & 7) + ((kk >> 4) << 3);
        xp[base + (size_t)kt * 512 + half * 256 + r * 16 + slot] = f2bf(xn);
    }
}

// ---------------------------------------------------------------------------
// Tiled bf16 WMMA GEMM: C[M,N] = A[M,K] @ B[K,N] (+bias, epilogue).
// Wave tile 32(M) x 64(N): 8 accumulators, 2 A + 4 B fragments per K-step
// (~21 flop/byte from L2). Block = 8 waves (4x2) -> 128x128 block tile.
// mode 0: out = z + bias                 (x_up)
// mode 1: out = sigmoid(silu(z + bias))  (delta & gate)
// mode 2: out = z + bias + resid         (down proj + residual)
// ---------------------------------------------------------------------------
__global__ __launch_bounds__(256) void gemm_wmma(const unsigned short* __restrict__ Ap,
                                                 const unsigned short* __restrict__ Bp,
                                                 const float* __restrict__ bias,
                                                 float* __restrict__ Cout,
                                                 const float* __restrict__ resid,
                                                 int K, int N, int mode) {
    const int lane = threadIdx.x & 31;
    const int w    = threadIdx.x >> 5;
    const int m0   = blockIdx.y * 128 + (w >> 1) * 32;
    const int n0   = blockIdx.x * 128 + (w & 1) * 64;
    const int Kt   = K >> 5;   // K tiles of 32
    const int Nt   = N >> 4;   // N tiles of 16

    const unsigned short* ap0 = Ap + (((size_t)(m0 >> 4)) * Kt << 9) + lane * 16;
    const unsigned short* ap1 = ap0 + ((size_t)Kt << 9);
    const unsigned short* bp  = Bp + (((size_t)(n0 >> 4)) << 9) + lane * 16;
    const size_t bstride = (size_t)Nt << 9;

    v8f acc[2][4];
    #pragma unroll
    for (int i = 0; i < 2; ++i)
        #pragma unroll
        for (int j = 0; j < 4; ++j) acc[i][j] = (v8f){};

    for (int kt = 0; kt < Kt; ++kt) {
        if (kt + 8 < Kt) {
            __builtin_prefetch(ap0 + (((size_t)kt + 8) << 9), 0, 1);
            __builtin_prefetch(ap1 + (((size_t)kt + 8) << 9), 0, 1);
            __builtin_prefetch(bp + ((size_t)(kt + 8)) * bstride, 0, 1);
        }
        v16bf af[2], bf[4];
        af[0] = *reinterpret_cast<const v16bf*>(ap0 + ((size_t)kt << 9));
        af[1] = *reinterpret_cast<const v16bf*>(ap1 + ((size_t)kt << 9));
        #pragma unroll
        for (int j = 0; j < 4; ++j)
            bf[j] = *reinterpret_cast<const v16bf*>(bp + (size_t)kt * bstride + (size_t)j * 512);
        #pragma unroll
        for (int i = 0; i < 2; ++i)
            #pragma unroll
            for (int j = 0; j < 4; ++j)
                acc[i][j] = __builtin_amdgcn_wmma_f32_16x16x32_bf16(
                    false, af[i], false, bf[j], (short)0, acc[i][j], false, false);
    }

    const int mrow = (lane >> 4) * 8;   // C layout: lanes 16-31 hold M=r+8
    const int nc   = lane & 15;
    #pragma unroll
    for (int i = 0; i < 2; ++i) {
        #pragma unroll
        for (int j = 0; j < 4; ++j) {
            #pragma unroll
            for (int r = 0; r < 8; ++r) {
                int   mi = m0 + i * 16 + mrow + r;
                int   ni = n0 + j * 16 + nc;
                float cv = acc[i][j][r] + bias[ni];
                float o;
                if (mode == 0)      o = cv;
                else if (mode == 1) { float sl = cv * sigmf(cv); o = sigmf(sl); }
                else                o = cv + resid[(size_t)mi * N + ni];
                Cout[(size_t)mi * N + ni] = o;
            }
        }
    }
}

// ---------------------------------------------------------------------------
// Selective scan, chunk-parallel (linear recurrence in h):
//   h_t = e_t * h_{t-1} + u_t * B,  e_t = exp(dt_t * A)  (diagonal per state)
// Pass 1: per (b,ch,chunk) from h=0: local end-state hend and decay prod P.
// Pass 2: per (b,ch): sequential combine of NCH chunks -> H_start per chunk.
// Pass 3: per (b,ch,chunk): rerun chunk from H_start, emit y (packed bf16).
// ---------------------------------------------------------------------------
__global__ __launch_bounds__(256) void scan_pass1(const float* __restrict__ xup,
                                                  const float* __restrict__ dg,
                                                  const float* __restrict__ Am,
                                                  const float* __restrict__ Bm,
                                                  float* __restrict__ hend,
                                                  float* __restrict__ Pprod) {
    int blk = blockIdx.x;
    int ch  = ((blk & 7) << 8) + threadIdx.x;   // 0..2047
    int c   = (blk >> 3) & (NCH - 1);
    int b   = blk >> 7;                         // /(8*NCH)
    float a[D_STATE], bb[D_STATE], h[D_STATE], P[D_STATE];
    #pragma unroll
    for (int s = 0; s < D_STATE; ++s) {
        a[s]  = Am[(size_t)ch * D_STATE + s];
        bb[s] = Bm[(size_t)ch * D_STATE + s];
        h[s] = 0.f; P[s] = 1.f;
    }
    int t0 = c * CHUNK;
    for (int i = 0; i < CHUNK; ++i) {
        size_t row = (size_t)b * SEQ + t0 + i;
        float u  = xup[row * D_INNER + ch];
        float dt = dg[row * (2 * D_INNER) + ch];
        #pragma unroll
        for (int s = 0; s < D_STATE; ++s) {
            float e = __expf(dt * a[s]);
            h[s] = h[s] * e + u * bb[s];
            P[s] *= e;
        }
    }
    size_t base = (((size_t)b * NCH + c) * D_STATE) * D_INNER + ch;
    #pragma unroll
    for (int s = 0; s < D_STATE; ++s) {
        hend [base + (size_t)s * D_INNER] = h[s];
        Pprod[base + (size_t)s * D_INNER] = P[s];
    }
}

__global__ __launch_bounds__(256) void scan_pass2(const float* __restrict__ hend,
                                                  const float* __restrict__ Pprod,
                                                  float* __restrict__ Hstart) {
    int b  = blockIdx.x >> 3;
    int ch = ((blockIdx.x & 7) << 8) + threadIdx.x;
    float h[D_STATE];
    #pragma unroll
    for (int s = 0; s < D_STATE; ++s) h[s] = 0.f;
    for (int c = 0; c < NCH; ++c) {
        size_t base = (((size_t)b * NCH + c) * D_STATE) * D_INNER + ch;
        #pragma unroll
        for (int s = 0; s < D_STATE; ++s) {
            size_t o = base + (size_t)s * D_INNER;
            Hstart[o] = h[s];
            h[s] = Pprod[o] * h[s] + hend[o];
        }
    }
}

__global__ __launch_bounds__(256) void scan_pass3(const float* __restrict__ xup,
                                                  const float* __restrict__ dg,
                                                  const float* __restrict__ Am,
                                                  const float* __restrict__ Bm,
                                                  const float* __restrict__ Cm,
                                                  const float* __restrict__ Dv,
                                                  const float* __restrict__ Hstart,
                                                  unsigned short* __restrict__ ypk) {
    int blk = blockIdx.x;
    int ch  = ((blk & 7) << 8) + threadIdx.x;
    int c   = (blk >> 3) & (NCH - 1);
    int b   = blk >> 7;
    float a[D_STATE], bb[D_STATE], cc[D_STATE], h[D_STATE];
    size_t hbase = (((size_t)b * NCH + c) * D_STATE) * D_INNER + ch;
    #pragma unroll
    for (int s = 0; s < D_STATE; ++s) {
        a[s]  = Am[(size_t)ch * D_STATE + s];
        bb[s] = Bm[(size_t)ch * D_STATE + s];
        cc[s] = Cm[(size_t)ch * D_STATE + s];
        h[s]  = Hstart[hbase + (size_t)s * D_INNER];
    }
    float dval = Dv[ch];

    int kt = ch >> 5, kk = ch & 31;
    int half  = (kk >> 3) & 1;
    int slot  = (kk & 7) + ((kk >> 4) << 3);
    int fixed = kt * 512 + half * 256 + slot;
    const int Ktiles = D_INNER / 32;   // 64

    int t0 = c * CHUNK;
    for (int i = 0; i < CHUNK; ++i) {
        size_t row = (size_t)b * SEQ + t0 + i;
        float u  = xup[row * D_INNER + ch];
        float dt = dg[row * (2 * D_INNER) + ch];             // sigmoid(silu(.))
        float g  = dg[row * (2 * D_INNER) + D_INNER + ch];   // sigmoid(silu(.))
        float y = 0.f;
        #pragma unroll
        for (int s = 0; s < D_STATE; ++s) {
            h[s] = h[s] * __expf(dt * a[s]) + u * bb[s];
            y += h[s] * cc[s];
        }
        y = (y + u * dval) * g;
        int m = (int)row;
        size_t off = (((size_t)(m >> 4) * Ktiles) << 9) + (size_t)((m & 15) * 16) + fixed;
        ypk[off] = f2bf(y);
    }
}

// ---------------------------------------------------------------------------
extern "C" void kernel_launch(void* const* d_in, const int* in_sizes, int n_in,
                              void* d_out, int out_size, void* d_ws, size_t ws_size,
                              hipStream_t stream) {
    (void)in_sizes; (void)n_in; (void)out_size; (void)ws_size;
    const float* x        = (const float*)d_in[0];
    const float* W_up     = (const float*)d_in[1];
    const float* b_up     = (const float*)d_in[2];
    const float* W_params = (const float*)d_in[3];
    const float* b_params = (const float*)d_in[4];
    const float* W_down   = (const float*)d_in[5];
    const float* b_down   = (const float*)d_in[6];
    const float* A        = (const float*)d_in[7];
    const float* B        = (const float*)d_in[8];
    const float* C        = (const float*)d_in[9];
    const float* D        = (const float*)d_in[10];
    const float* ln_gamma = (const float*)d_in[11];
    const float* ln_beta  = (const float*)d_in[12];
    float* out = (float*)d_out;

    char* base = (char*)d_ws;
    size_t off = 0;
    auto alloc = [&](size_t bytes) { char* p = base + off; off += (bytes + 255) & ~(size_t)255; return p; };
    unsigned short* xnp   = (unsigned short*)alloc((size_t)MROWS * D_MODEL * 2);
    unsigned short* wupp  = (unsigned short*)alloc((size_t)D_MODEL * D_INNER * 2);
    unsigned short* wparp = (unsigned short*)alloc((size_t)D_MODEL * 2 * D_INNER * 2);
    unsigned short* wdnp  = (unsigned short*)alloc((size_t)D_INNER * D_MODEL * 2);
    float*          xup   = (float*)alloc((size_t)MROWS * D_INNER * 4);
    float*          dg    = (float*)alloc((size_t)MROWS * 2 * D_INNER * 4);
    unsigned short* ypk   = (unsigned short*)alloc((size_t)MROWS * D_INNER * 2);
    float*          hend  = (float*)alloc((size_t)BATCH * NCH * D_STATE * D_INNER * 4);
    float*          pprod = (float*)alloc((size_t)BATCH * NCH * D_STATE * D_INNER * 4);
    float*          hst   = (float*)alloc((size_t)BATCH * NCH * D_STATE * D_INNER * 4);

    // 1) weight repacks (fp32 -> bf16 fragment layout)
    repack_w<<<(D_MODEL * D_INNER + 255) / 256, 256, 0, stream>>>(W_up, wupp, D_MODEL, D_INNER);
    repack_w<<<(D_MODEL * 2 * D_INNER + 255) / 256, 256, 0, stream>>>(W_params, wparp, D_MODEL, 2 * D_INNER);
    repack_w<<<(D_INNER * D_MODEL + 255) / 256, 256, 0, stream>>>(W_down, wdnp, D_INNER, D_MODEL);

    // 2) LayerNorm -> packed bf16 A fragments
    ln_pack<<<MROWS, 256, 0, stream>>>(x, ln_gamma, ln_beta, xnp);

    // 3) x_up = xn @ W_up + b_up
    gemm_wmma<<<dim3(D_INNER / 128, MROWS / 128), 256, 0, stream>>>(
        xnp, wupp, b_up, xup, nullptr, D_MODEL, D_INNER, 0);

    // 4) delta/gate = sigmoid(silu(xn @ W_params + b_params))
    gemm_wmma<<<dim3(2 * D_INNER / 128, MROWS / 128), 256, 0, stream>>>(
        xnp, wparp, b_params, dg, nullptr, D_MODEL, 2 * D_INNER, 1);

    // 5) chunk-parallel selective scan -> packed bf16 y
    scan_pass1<<<BATCH * NCH * (D_INNER / 256), 256, 0, stream>>>(xup, dg, A, B, hend, pprod);
    scan_pass2<<<BATCH * (D_INNER / 256), 256, 0, stream>>>(hend, pprod, hst);
    scan_pass3<<<BATCH * NCH * (D_INNER / 256), 256, 0, stream>>>(xup, dg, A, B, C, D, hst, ypk);

    // 6) out = y @ W_down + b_down + x
    gemm_wmma<<<dim3(D_MODEL / 128, MROWS / 128), 256, 0, stream>>>(
        ypk, wdnp, b_down, out, x, D_INNER, D_MODEL, 2);
}
